// Corr_71957882077190
// MI455X (gfx1250) — compile-verified, matrix operand order checked by reading, not run on hardware
//
#include <hip/hip_runtime.h>

// CDNA5 / gfx1250: wave32, WMMA 16x16x32 bf16 -> f32
typedef __attribute__((ext_vector_type(16))) __bf16 v16bf;
typedef __attribute__((ext_vector_type(8)))  float  v8f;
typedef __attribute__((ext_vector_type(4)))  float  v4f;

#define B_TOTAL   512
#define C_TOTAL   256
#define KXD       6
#define KZD       22
#define ZPIX      (KZD * KZD)         // 484
#define XPIX      (KXD * KXD)         // 36
#define OUT_W     17                  // 22 - 6 + 1
#define OUT_N     289                 // 17*17
#define C_CHUNK   32
#define N_CPASS   (C_TOTAL / C_CHUNK) // 8
#define ZSTRIDE   40                  // C_CHUNK + 8 pad halfs -> 80B rows (16B aligned, bank-spread)
#define XSTRIDE   40
#define NTILES    19                  // ceil(289/16)
#define WAVES     8
#define THREADS   256
#define MAXT      3                   // tiles per wave (19/8 rounded up)

// f32 -> bf16 round-to-nearest-even (raw bits)
__device__ __forceinline__ unsigned short f2bf(float f) {
    unsigned u = __float_as_uint(f);
    unsigned r = u + 0x7fffu + ((u >> 16) & 1u);
    return (unsigned short)(r >> 16);
}

union AFrag {
    v16bf v;
    uint4 u[2];
};

// Compile-time LDS offsets (halfs) for shift (i,j) = (ij/6, ij%6)
#define ZOFF(ij) ((((ij) / KXD) * KZD + ((ij) % KXD)) * ZSTRIDE)
#define XOFF(ij) ((ij) * XSTRIDE)

__global__ __launch_bounds__(THREADS)
void corr_wmma_kernel(const float* __restrict__ x,
                      const float* __restrict__ z,
                      float* __restrict__ out)
{
    // zT[pixel][channel] and x2[ij][channel], bf16, padded rows for alignment/banking
    __shared__ __align__(16) unsigned short zlds[ZPIX * ZSTRIDE]; // 38,720 B
    __shared__ __align__(16) unsigned short xlds[XPIX * XSTRIDE]; //  2,880 B

    const int b     = blockIdx.x;
    const int tid   = threadIdx.x;
    const int lane  = tid & 31;
    const int mlane = lane & 15;          // M row within the 16x16 tile
    const int koff  = (lane >> 4) << 4;   // K 0-15 for lanes 0-15, K 16-31 for lanes 16-31
    // Scalarized wave id: keeps tile guards on SALU branches (EXEC stays all-1s for WMMA)
    const int wv    = __builtin_amdgcn_readfirstlane(tid >> 5);

    // Per-wave static tile assignment; precompute im2col base pixel per lane.
    int zb[MAXT];
#pragma unroll
    for (int tt = 0; tt < MAXT; ++tt) {
        int t = wv + tt * WAVES;
        int pos = t * 16 + mlane;
        if (pos >= OUT_N) pos = OUT_N - 1;   // clamp pad lanes to a valid pixel
        int p = pos / OUT_W;
        int q = pos - p * OUT_W;
        zb[tt] = p * KZD + q;                // top-left pixel of the window
    }

    v8f acc[MAXT] = {};

    const size_t zbatch = (size_t)b * C_TOTAL * ZPIX;
    const size_t xbatch = (size_t)b * C_TOTAL * XPIX;

    for (int cp = 0; cp < N_CPASS; ++cp) {
        const int c0 = cp * C_CHUNK;

        // ---- Stage z chunk: 32 contiguous channels = 15488 f32, 128-bit NT loads ----
        {
            const v4f* zf4 = (const v4f*)(z + zbatch + (size_t)c0 * ZPIX);
            for (int vv = tid; vv < (C_CHUNK * ZPIX) / 4; vv += THREADS) {
                v4f f = __builtin_nontemporal_load(&zf4[vv]);
                const int e = vv * 4;
#pragma unroll
                for (int k = 0; k < 4; ++k) {
                    int ee  = e + k;
                    int c   = ee / ZPIX;
                    int pix = ee - c * ZPIX;
                    zlds[pix * ZSTRIDE + c] = f2bf(f[k]);
                }
            }
        }
        // ---- Stage x chunk: 32 channels x 36 = 1152 f32 ----
        {
            const v4f* xf4 = (const v4f*)(x + xbatch + (size_t)c0 * XPIX);
            for (int vv = tid; vv < (C_CHUNK * XPIX) / 4; vv += THREADS) {
                v4f f = __builtin_nontemporal_load(&xf4[vv]);
                const int e = vv * 4;
#pragma unroll
                for (int k = 0; k < 4; ++k) {
                    int ee = e + k;
                    int c  = ee / XPIX;
                    int ij = ee - c * XPIX;
                    xlds[ij * XSTRIDE + c] = f2bf(f[k]);
                }
            }
        }
        // Prefetch next channel chunk of z into cache while this chunk is computed on
        if (cp + 1 < N_CPASS) {
            const char* nz = (const char*)(z + zbatch + (size_t)(c0 + C_CHUNK) * ZPIX);
            if (tid < (C_CHUNK * ZPIX * 4) / 256)        // one 256B line per thread
                __builtin_prefetch(nz + tid * 256, 0, 1);
        }
        __syncthreads();

        // ---- WMMA over the 36 kernel shifts, double-buffered fragments ----
#define LDA(f, idx) do { (f).u[0] = *(const uint4*)&zlds[(idx)];      \
                         (f).u[1] = *(const uint4*)&zlds[(idx) + 8]; } while (0)
#define LDB(f, idx) do { (f).u[0] = *(const uint4*)&xlds[(idx)];      \
                         (f).u[1] = *(const uint4*)&xlds[(idx) + 8]; } while (0)
#pragma unroll
        for (int tt = 0; tt < MAXT; ++tt) {
            if (wv + tt * WAVES >= NTILES) continue;     // scalar (SALU) branch
            const int aB = zb[tt] * ZSTRIDE + koff;
            const int bB = koff;
            AFrag a0, b0, a1, b1;
            LDA(a0, aB + ZOFF(0));
            LDB(b0, bB + XOFF(0));
            v8f c = acc[tt];
#pragma unroll
            for (int u = 0; u < XPIX; u += 2) {
                LDA(a1, aB + ZOFF(u + 1));
                LDB(b1, bB + XOFF(u + 1));
                c = __builtin_amdgcn_wmma_f32_16x16x32_bf16(
                        false, a0.v, false, b0.v, (short)0, c, false, false);
                if (u + 2 < XPIX) {
                    LDA(a0, aB + ZOFF(u + 2));
                    LDB(b0, bB + XOFF(u + 2));
                }
                c = __builtin_amdgcn_wmma_f32_16x16x32_bf16(
                        false, a1.v, false, b1.v, (short)0, c, false, false);
            }
            acc[tt] = c;
        }
        __syncthreads();
    }

    // All 16 D columns are identical; lane 0 holds M=0..7, lane 16 holds M=8..15.
    float* ob = out + (size_t)b * OUT_N;
#pragma unroll
    for (int tt = 0; tt < MAXT; ++tt) {
        if (wv + tt * WAVES >= NTILES) continue;
        const int base = (wv + tt * WAVES) * 16;
        if (lane == 0) {
#pragma unroll
            for (int m = 0; m < 8; ++m)
                if (base + m < OUT_N) ob[base + m] = acc[tt][m];
        } else if (lane == 16) {
#pragma unroll
            for (int m = 0; m < 8; ++m)
                if (base + 8 + m < OUT_N) ob[base + 8 + m] = acc[tt][m];
        }
    }
}

extern "C" void kernel_launch(void* const* d_in, const int* in_sizes, int n_in,
                              void* d_out, int out_size, void* d_ws, size_t ws_size,
                              hipStream_t stream) {
    const float* x = (const float*)d_in[0];   // [512,256,6,6]
    const float* z = (const float*)d_in[1];   // [512,256,22,22]
    float* out = (float*)d_out;               // [512,1,17,17]
    (void)in_sizes; (void)n_in; (void)out_size; (void)d_ws; (void)ws_size;
    hipLaunchKernelGGL(corr_wmma_kernel, dim3(B_TOTAL), dim3(THREADS), 0, stream,
                       x, z, out);
}
